// Generator_36842229465601
// MI455X (gfx1250) — matrix operand order repaired and checked by reading, not compiled, not run on hardware
//
#include <hip/hip_runtime.h>

// ---------------------------------------------------------------------------
// MI455X / gfx1250 (CDNA5, wave32) implementation.
//
//  compute-bound (~500 FLOP/byte) -> f16 WMMA (v_wmma_f32_16x16x32_f16),
//  f32 accumulation, weights resident in LDS (staged via the Tensor Data
//  Mover: tensor_load_to_lds + s_wait_tensorcnt), per-wave 16-row tiles.
//
//  ws layout (as _Float16 elements):
//    [0      .. 65536)   lstm_kernel  f16 (128 x 512, row major)
//    [65536  .. 131072)  lstm_rec     f16 (128 x 512, row major)
//    [131072 .. 147456)  W1 f16 (128 x 128)
//    [147456 .. 163840)  W2 f16
//    [163840 .. 180224)  W3 f16
//    [180224 .. 180608)  att f16 (3 x 128)
//    [180736 .. +B*3*128) hidden states f16, row = b*3 + t   (~101 MB total)
// ---------------------------------------------------------------------------

typedef _Float16 v8h  __attribute__((ext_vector_type(8)));
typedef _Float16 v16h __attribute__((ext_vector_type(16)));
typedef float    v8f  __attribute__((ext_vector_type(8)));
typedef unsigned int u32x4 __attribute__((ext_vector_type(4)));
typedef int      i32x4 __attribute__((ext_vector_type(4)));
typedef int      i32x8 __attribute__((ext_vector_type(8)));

#define BATCH 131072
#define HID   128
#define FEAT  128

// LSTM kernel LDS byte offsets
#define SM_WK        0
#define SM_WR        131072
#define SM_BIAS      262144
#define SM_ATT       264192
#define SM_SWAP      264960
#define SM_LSTM_SZ   297728
// MLP kernel LDS byte offsets
#define SM2_W        0
#define SM2_BIAS     98304
#define SM2_SWAP     99840
#define SM2_SZ       132608

static __device__ __forceinline__ v16h cat16(v8h a, v8h b) {
  return __builtin_shufflevector(a, b, 0,1,2,3,4,5,6,7,8,9,10,11,12,13,14,15);
}
static __device__ __forceinline__ v8f wmma16(v16h a, v16h b, v8f c) {
  return __builtin_amdgcn_wmma_f32_16x16x32_f16(false, a, false, b, (short)0, c,
                                                false, false);
}
// B fragment: lane l holds column N = l%16; p points at row 8*(l/16) of the
// 32-row K-chunk; slots j<8 = rows j, slots j>=8 = rows 16+(j-8) (stride ld).
static __device__ __forceinline__ v16h ldfragB(const _Float16* p, int ld) {
  v16h v;
#pragma unroll
  for (int s = 0; s < 8; ++s) {
    v[s]     = p[s * ld];
    v[s + 8] = p[(16 + s) * ld];
  }
  return v;
}
static __device__ __forceinline__ float sigmoidf_(float x) {
  return 1.0f / (1.0f + __expf(-x));
}
static __device__ __forceinline__ v8f splat8(float v) {
  v8f r;
#pragma unroll
  for (int i = 0; i < 8; ++i) r[i] = v;
  return r;
}

// ---------------------------------------------------------------------------
// TDM: DMA a contiguous block (n_u64 * 8 bytes) from global memory into LDS.
// Descriptor per CDNA5 ISA:
//   group0: count=1 | lds_addr[63:32] | global_addr[120:64] | type=2
//   group1: data_size=3 (8B), 2D tensor: dim0 x rows, tile == tensor,
//           tensor_dim0_stride = dim0.  Tracked with TENSORcnt.
// ---------------------------------------------------------------------------
static __device__ __forceinline__ void tdm_load_1d(const void* gsrc,
                                                   unsigned lds_byte_off,
                                                   unsigned n_u64) {
  const unsigned dim0 = (n_u64 >= 4096u) ? 4096u : n_u64;
  const unsigned rows = n_u64 / dim0;
  const unsigned long long ga = (unsigned long long)(uintptr_t)gsrc;
  u32x4 g0;
  g0[0] = 1u;                                              // count=1 (user D#)
  g0[1] = lds_byte_off;                                    // lds_addr
  g0[2] = (unsigned)(ga & 0xffffffffu);                    // global_addr[31:0]
  g0[3] = (unsigned)((ga >> 32) & 0x01ffffffu) | (2u << 30);  // ga[56:32]|type=2
  i32x8 g1;
  g1[0] = (int)(3u << 16);                                 // data_size = 8B
  g1[1] = (int)((dim0 & 0xffffu) << 16);                   // tensor_dim0 lo16
  g1[2] = (int)((dim0 >> 16) | ((rows & 0xffffu) << 16));  // dim0 hi | dim1 lo
  g1[3] = (int)((rows >> 16) | ((dim0 & 0xffffu) << 16));  // dim1 hi | tile_dim0
  g1[4] = (int)(rows & 0xffffu);                           // tile_dim1 (dim2=0)
  g1[5] = (int)dim0;                                       // dim0_stride[31:0]
  g1[6] = 0;
  g1[7] = 0;
  const i32x4 z4 = {0, 0, 0, 0};
#if __clang_major__ >= 23
  const i32x8 z8 = {0, 0, 0, 0, 0, 0, 0, 0};
  __builtin_amdgcn_tensor_load_to_lds(g0, g1, z4, z4, z8, 0);
#else
  __builtin_amdgcn_tensor_load_to_lds(g0, g1, z4, z4, 0);
#endif
}

__global__ void cvt_f32_to_f16(const float* __restrict__ s,
                               _Float16* __restrict__ d, int n) {
  int i = blockIdx.x * blockDim.x + threadIdx.x;
  if (i < n) d[i] = (_Float16)s[i];
}

// ---------------------------------------------------------------------------
// Phase 1: fused attention-scale + 3-step LSTM.  8 waves/block, 16 rows/wave.
// ---------------------------------------------------------------------------
__global__ void __launch_bounds__(256)
lstm_fused(const float* __restrict__ ctx,       // (B,3,H) f32
           const _Float16* __restrict__ gwk,    // 128x512 f16
           const _Float16* __restrict__ gwr,    // 128x512 f16
           const _Float16* __restrict__ gatt,   // 3x128 f16
           const float* __restrict__ gbias,     // 512 f32
           _Float16* __restrict__ hout)         // (B*3,128) f16
{
  extern __shared__ char smem[];
  _Float16* sWk  = (_Float16*)(smem + SM_WK);
  _Float16* sWr  = (_Float16*)(smem + SM_WR);
  float*    sB   = (float*)   (smem + SM_BIAS);
  _Float16* sAtt = (_Float16*)(smem + SM_ATT);
  _Float16* sSw  = (_Float16*)(smem + SM_SWAP);

  const int tid = threadIdx.x;
  // ---- TDM staging: wave 0 issues 4 tensor DMAs, waits on TENSORcnt ----
  if (tid < 32) {
    const unsigned smbase = (unsigned)(uintptr_t)smem;
    tdm_load_1d(gwk,   smbase + SM_WK,   131072u / 8u);
    tdm_load_1d(gwr,   smbase + SM_WR,   131072u / 8u);
    tdm_load_1d(gbias, smbase + SM_BIAS, 2048u / 8u);
    tdm_load_1d(gatt,  smbase + SM_ATT,  768u / 8u);
    __builtin_amdgcn_s_wait_tensorcnt(0);
  }
  __syncthreads();

  const int w    = tid >> 5;
  const int lane = tid & 31;
  const int lm   = lane & 15;   // A rows / B cols / C cols
  const int lh   = lane >> 4;   // lane half
  const long rb  = (long)blockIdx.x * 128 + (long)w * 16;

  const float* rowp = ctx + ((rb + lm) * 3 + 2) * HID;  // last_hidden row

  // cell state, 8 16x16 C-layout tiles (f32)
  v8f c8[8];
#pragma unroll
  for (int nt = 0; nt < 8; ++nt) c8[nt] = splat8(0.0f);

  // h in A-fragment layout, zero-initialized (h0 = 0)
  v16h hz;
#pragma unroll
  for (int j = 0; j < 16; ++j) hz[j] = (_Float16)0.0f;
  v16h hf[4] = {hz, hz, hz, hz};

  for (int t = 0; t < 3; ++t) {
    // x_t = last_hidden * att_t, built directly in A-fragment layout
    v16h xf[4];
#pragma unroll
    for (int kc = 0; kc < 4; ++kc) {
      float4 a0 = *(const float4*)(rowp + kc * 32 + lh * 8);
      float4 a1 = *(const float4*)(rowp + kc * 32 + lh * 8 + 4);
      float4 b0 = *(const float4*)(rowp + kc * 32 + 16 + lh * 8);
      float4 b1 = *(const float4*)(rowp + kc * 32 + 16 + lh * 8 + 4);
      v8h at0 = *(const v8h*)(sAtt + t * HID + kc * 32 + lh * 8);
      v8h at1 = *(const v8h*)(sAtt + t * HID + kc * 32 + 16 + lh * 8);
      float r0[8] = {a0.x, a0.y, a0.z, a0.w, a1.x, a1.y, a1.z, a1.w};
      float r1[8] = {b0.x, b0.y, b0.z, b0.w, b1.x, b1.y, b1.z, b1.w};
      v16h v;
#pragma unroll
      for (int j = 0; j < 8; ++j) {
        v[j]     = (_Float16)r0[j] * at0[j];
        v[j + 8] = (_Float16)r1[j] * at1[j];
      }
      xf[kc] = v;
    }

#pragma unroll
    for (int nt = 0; nt < 8; ++nt) {
      // gate accumulators i,f,g,o (keras order), bias-initialized
      v8f acc[4];
#pragma unroll
      for (int g = 0; g < 4; ++g) acc[g] = splat8(sB[g * FEAT + nt * 16 + lm]);

#pragma unroll
      for (int kc = 0; kc < 4; ++kc) {
        const int krow = kc * 32 + 8 * lh;
        const _Float16* wkp = sWk + krow * 512;
        const _Float16* wrp = sWr + krow * 512;
#pragma unroll
        for (int g = 0; g < 4; ++g) {
          acc[g] = wmma16(xf[kc], ldfragB(wkp + g * FEAT + nt * 16 + lm, 512), acc[g]);
          acc[g] = wmma16(hf[kc], ldfragB(wrp + g * FEAT + nt * 16 + lm, 512), acc[g]);
        }
      }

      // elementwise gate math on f32 accumulators (C layout)
      v8f hvals;
#pragma unroll
      for (int e = 0; e < 8; ++e) {
        float ig = sigmoidf_(acc[0][e]);
        float fg = sigmoidf_(acc[1][e]);
        float gg = tanhf(acc[2][e]);
        float og = sigmoidf_(acc[3][e]);
        float cn = fg * c8[nt][e] + ig * gg;
        c8[nt][e] = cn;
        hvals[e] = og * tanhf(cn);
      }

      // store h tile (C layout): LDS bounce for A-relayout + global f16
#pragma unroll
      for (int e = 0; e < 8; ++e) {
        const int m = e + lh * 8;
        const _Float16 hv = (_Float16)hvals[e];
        sSw[w * 2048 + m * 128 + nt * 16 + lm] = hv;
        hout[((rb + m) * 3 + t) * (long)FEAT + nt * 16 + lm] = hv;
      }
    }

    // reload h as A-fragments (same-wave LDS ops are in order)
#pragma unroll
    for (int kc = 0; kc < 4; ++kc) {
      const _Float16* rp = sSw + w * 2048 + lm * 128 + kc * 32;
      hf[kc] = cat16(*(const v8h*)(rp + lh * 8),
                     *(const v8h*)(rp + 16 + lh * 8));
    }
  }
}

// ---------------------------------------------------------------------------
// Phase 2: 3x (dense 128x128 + ReLU) over the (B*3, 128) hidden matrix.
// ---------------------------------------------------------------------------
__global__ void __launch_bounds__(256)
mlp3(const _Float16* __restrict__ hin,   // (B*3,128) f16
     const _Float16* __restrict__ gw1,
     const _Float16* __restrict__ gw2,
     const _Float16* __restrict__ gw3,
     const float* __restrict__ gb1,
     const float* __restrict__ gb2,
     const float* __restrict__ gb3,
     float* __restrict__ out)            // (B*3,128) f32
{
  extern __shared__ char smem[];
  _Float16* sW  = (_Float16*)(smem + SM2_W);
  float*    sB  = (float*)   (smem + SM2_BIAS);
  _Float16* sSw = (_Float16*)(smem + SM2_SWAP);

  const int tid = threadIdx.x;
  // ---- TDM staging: wave 0 issues 6 tensor DMAs, waits on TENSORcnt ----
  if (tid < 32) {
    const unsigned smbase = (unsigned)(uintptr_t)smem;
    tdm_load_1d(gw1, smbase + SM2_W,          32768u / 8u);
    tdm_load_1d(gw2, smbase + SM2_W + 32768u, 32768u / 8u);
    tdm_load_1d(gw3, smbase + SM2_W + 65536u, 32768u / 8u);
    tdm_load_1d(gb1, smbase + SM2_BIAS,         512u / 8u);
    tdm_load_1d(gb2, smbase + SM2_BIAS + 512u,  512u / 8u);
    tdm_load_1d(gb3, smbase + SM2_BIAS + 1024u, 512u / 8u);
    __builtin_amdgcn_s_wait_tensorcnt(0);
  }
  __syncthreads();

  const int w    = tid >> 5;
  const int lane = tid & 31;
  const int lm   = lane & 15;
  const int lh   = lane >> 4;
  const long rb  = (long)blockIdx.x * 128 + (long)w * 16;  // row in (B*3)

  // load activations as A-fragments (f16, row major, 2x16B per K-chunk)
  v16h yf[4];
  const _Float16* rp = hin + (rb + lm) * (long)FEAT;
#pragma unroll
  for (int kc = 0; kc < 4; ++kc) {
    yf[kc] = cat16(*(const v8h*)(rp + kc * 32 + lh * 8),
                   *(const v8h*)(rp + kc * 32 + 16 + lh * 8));
  }

  for (int layer = 0; layer < 3; ++layer) {
    const _Float16* wl = sW + layer * 16384;
    const float*    bl = sB + layer * 128;
#pragma unroll
    for (int nt = 0; nt < 8; ++nt) {
      v8f acc = splat8(bl[nt * 16 + lm]);
#pragma unroll
      for (int kc = 0; kc < 4; ++kc) {
        const int krow = kc * 32 + 8 * lh;
        acc = wmma16(yf[kc], ldfragB(wl + krow * FEAT + nt * 16 + lm, FEAT), acc);
      }
#pragma unroll
      for (int e = 0; e < 8; ++e) {
        float v = acc[e] > 0.0f ? acc[e] : 0.0f;
        const int m = e + lh * 8;
        if (layer == 2) {
          out[(rb + m) * (long)FEAT + nt * 16 + lm] = v;
        } else {
          sSw[w * 2048 + m * 128 + nt * 16 + lm] = (_Float16)v;
        }
      }
    }
    if (layer < 2) {
#pragma unroll
      for (int kc = 0; kc < 4; ++kc) {
        const _Float16* p = sSw + w * 2048 + lm * 128 + kc * 32;
        yf[kc] = cat16(*(const v8h*)(p + lh * 8),
                       *(const v8h*)(p + 16 + lh * 8));
      }
    }
  }
}

extern "C" void kernel_launch(void* const* d_in, const int* in_sizes, int n_in,
                              void* d_out, int out_size, void* d_ws, size_t ws_size,
                              hipStream_t stream) {
  const float* ctx  = (const float*)d_in[0];
  const float* a1   = (const float*)d_in[1];
  const float* a2   = (const float*)d_in[2];
  const float* a3   = (const float*)d_in[3];
  const float* wk   = (const float*)d_in[4];
  const float* wr   = (const float*)d_in[5];
  const float* bias = (const float*)d_in[6];
  const float* W1   = (const float*)d_in[7];
  const float* b1   = (const float*)d_in[8];
  const float* W2   = (const float*)d_in[9];
  const float* b2   = (const float*)d_in[10];
  const float* W3   = (const float*)d_in[11];
  const float* b3   = (const float*)d_in[12];

  _Float16* ws    = (_Float16*)d_ws;
  _Float16* wsWk  = ws;
  _Float16* wsWr  = ws + 65536;
  _Float16* wsW1  = ws + 131072;
  _Float16* wsW2  = ws + 147456;
  _Float16* wsW3  = ws + 163840;
  _Float16* wsAtt = ws + 180224;
  _Float16* wsH   = ws + 180736;  // (B*3,128) f16

  cvt_f32_to_f16<<<65536 / 256, 256, 0, stream>>>(wk, wsWk, 65536);
  cvt_f32_to_f16<<<65536 / 256, 256, 0, stream>>>(wr, wsWr, 65536);
  cvt_f32_to_f16<<<16384 / 256, 256, 0, stream>>>(W1, wsW1, 16384);
  cvt_f32_to_f16<<<16384 / 256, 256, 0, stream>>>(W2, wsW2, 16384);
  cvt_f32_to_f16<<<16384 / 256, 256, 0, stream>>>(W3, wsW3, 16384);
  cvt_f32_to_f16<<<1, 256, 0, stream>>>(a1, wsAtt, 128);
  cvt_f32_to_f16<<<1, 256, 0, stream>>>(a2, wsAtt + 128, 128);
  cvt_f32_to_f16<<<1, 256, 0, stream>>>(a3, wsAtt + 256, 128);

  (void)hipFuncSetAttribute(reinterpret_cast<const void*>(lstm_fused),
                            hipFuncAttributeMaxDynamicSharedMemorySize, SM_LSTM_SZ);
  (void)hipFuncSetAttribute(reinterpret_cast<const void*>(mlp3),
                            hipFuncAttributeMaxDynamicSharedMemorySize, SM2_SZ);

  lstm_fused<<<BATCH / 128, 256, SM_LSTM_SZ, stream>>>(ctx, wsWk, wsWr, wsAtt,
                                                       bias, wsH);
  mlp3<<<(BATCH * 3) / 128, 256, SM2_SZ, stream>>>(wsH, wsW1, wsW2, wsW3,
                                                   b1, b2, b3, (float*)d_out);
}
